// Customized_LSTM_47064251630154
// MI455X (gfx1250) — compile-verified, matrix operand order checked by reading, not run on hardware
//
#include <hip/hip_runtime.h>

// ---------------------------------------------------------------------------
// LSTM for MI455X (gfx1250): B=64, T=1024, D=256, UNITS=256, NUM_OUT=32
//   Kernel 1: xz = x @ U + b   (big parallel bf16-WMMA GEMM, off critical path)
//             stored TIME-MAJOR [T, B, 4U] so the serial recurrence reads a
//             contiguous 256 KB slice per step.
//   Kernel 2: persistent recurrence, 1024 steps, bf16 WMMA for h@W,
//             W in LDS, c-state in registers, atomic grid barrier per step.
// ---------------------------------------------------------------------------

#define BATCH  64
#define TLEN   1024
#define DIM    256
#define UNITS  256
#define GDIM   1024   // 4*UNITS
#define NOUT   32
#define NBLK_R 16     // recurrence blocks (16 units each)

typedef __attribute__((ext_vector_type(16))) __bf16        v16bf;
typedef __attribute__((ext_vector_type(8)))  float         v8f;
typedef __attribute__((ext_vector_type(8)))  unsigned int  v8u;

union FragCast { v8u u; v16bf b; };

__device__ __forceinline__ unsigned short f2bf(float x) {
    unsigned int u = __float_as_uint(x);
    unsigned int r = (u + 0x7FFFu + ((u >> 16) & 1u)) >> 16;   // RNE
    return (unsigned short)r;
}

// K offset held by VGPR j (0..7) for the 16-bit A/B fragment layout:
// lanes 0-15: VGPR0..3 -> K 0..7,  VGPR4..7 -> K 16..23
// lanes 16-31: +8
__device__ __forceinline__ int kmap(int j, int half) {
    return ((j < 4) ? (2 * j) : (16 + 2 * (j - 4))) + 8 * half;
}

__device__ __forceinline__ float fsigmoid(float x) {
    return 1.0f / (1.0f + __expf(-x));
}
__device__ __forceinline__ float ftanh(float x) {
    float e = __expf(2.0f * x);
    return (e - 1.0f) / (e + 1.0f);
}

// ------------------------- converters / init -------------------------------

__global__ void conv_x_kernel(const float* __restrict__ x,
                              unsigned short* __restrict__ xb, int n) {
    int i = blockIdx.x * blockDim.x + threadIdx.x;
    if (i < n) xb[i] = f2bf(x[i]);
}

// U: [K=256, N=1024] f32  ->  U_bf16T: [N=1024][K=256] (K contiguous pairs)
__global__ void conv_U_kernel(const float* __restrict__ U,
                              unsigned short* __restrict__ ub) {
    int i = blockIdx.x * blockDim.x + threadIdx.x;   // over 1024*256
    if (i < GDIM * DIM) {
        int n = i / DIM, k = i % DIM;
        ub[i] = f2bf(U[k * GDIM + n]);
    }
}

__global__ void init_state_kernel(unsigned short* __restrict__ hbuf,
                                  unsigned int* __restrict__ cnt) {
    int i = blockIdx.x * blockDim.x + threadIdx.x;
    for (int j = i; j < 2 * BATCH * UNITS; j += blockDim.x * gridDim.x)
        hbuf[j] = 0;
    if (i == 0) *cnt = 0u;
}

// ------------------- kernel 1: xz = x @ U + b (WMMA) -----------------------
// grid: (M/64)*(N/128) blocks, 256 threads = 8 waves.
// wave w: mt = w%4 (16-row tile), ngrp = w/4 -> 4 column tiles of 16.
// Output written time-major: xz[t][b][col].
__global__ __launch_bounds__(256)
void input_proj_kernel(const unsigned short* __restrict__ xb,   // [M,256] bf16
                       const unsigned short* __restrict__ ubT,  // [1024,256] bf16
                       const float* __restrict__ bias,          // [1024]
                       float* __restrict__ xz) {                // [T,B,1024] f32
    const int M = BATCH * TLEN;
    int blk  = blockIdx.x;
    int mblk = blk >> 3;          // 64-row block
    int nblk = blk & 7;           // 128-col block
    int w    = threadIdx.x >> 5;
    int lane = threadIdx.x & 31;
    int half = lane >> 4;
    int mlan = lane & 15;

    int row0 = mblk * 64 + (w & 3) * 16;         // wave's 16 rows (= b*T + t)
    int col0 = nblk * 128 + (w >> 2) * 64;       // wave's 64 cols
    if (row0 >= M) return;

    v8f acc[4] = {v8f(0.f), v8f(0.f), v8f(0.f), v8f(0.f)};

    const unsigned short* arow = xb + (row0 + mlan) * DIM;

#pragma unroll
    for (int ks = 0; ks < 8; ++ks) {
        // ---- A fragment: rows row0..row0+15, K = 32*ks..+31 ----
        FragCast fa;
#pragma unroll
        for (int j = 0; j < 8; ++j) {
            int kk = 32 * ks + kmap(j, half);
            fa.u[j] = *reinterpret_cast<const unsigned int*>(arow + kk);
        }
#pragma unroll
        for (int i = 0; i < 4; ++i) {
            int cg = col0 + i * 16 + mlan;       // this lane's column
            FragCast fb;
            const unsigned short* bcol = ubT + cg * DIM;
#pragma unroll
            for (int j = 0; j < 8; ++j) {
                int kk = 32 * ks + kmap(j, half);
                fb.u[j] = *reinterpret_cast<const unsigned int*>(bcol + kk);
            }
            acc[i] = __builtin_amdgcn_wmma_f32_16x16x32_bf16(
                false, fa.b, false, fb.b, (short)0, acc[i], false, false);
        }
    }

    // bias + store (time-major: row = b*TLEN + t  ->  xz[t*BATCH + b])
#pragma unroll
    for (int i = 0; i < 4; ++i) {
        int cg = col0 + i * 16 + mlan;
        float bv = bias[cg];
#pragma unroll
        for (int r = 0; r < 8; ++r) {
            int row  = row0 + r + 8 * half;
            int bidx = row / TLEN;               // batch (uniform within tile)
            int tt   = row % TLEN;               // time
            xz[((size_t)tt * BATCH + bidx) * GDIM + cg] = acc[i][r] + bv;
        }
    }
}

// --------------------- kernel 2: persistent recurrence ---------------------
// 16 blocks x 128 threads (4 waves). Block owns units [16*blk, 16*blk+16).
// Wave w owns batch rows [16w, 16w+16) and ALL 4 gates for those units.
__global__ __launch_bounds__(128)
void recurrence_kernel(const float* __restrict__ xz,        // [T, B, 1024]
                       const float* __restrict__ W,         // [256, 1024] f32
                       unsigned short* __restrict__ hbuf,   // 2 x [64,256] bf16
                       unsigned int* __restrict__ cnt,
                       float* __restrict__ out) {           // [64, 32, 256]
    __shared__ unsigned short Wlds[64 * DIM];   // [gate*16+u][k], 32 KB bf16

    const int tid   = threadIdx.x;
    const int w     = tid >> 5;
    const int lane  = tid & 31;
    const int half  = lane >> 4;
    const int nlan  = lane & 15;                // local unit
    const int ubase = blockIdx.x * 16;

    // --- load + transpose this block's 64 W columns into LDS (once) ---
    for (int i = tid; i < 64 * DIM; i += blockDim.x) {
        int lc = i / DIM;                        // 0..63 = gate*16 + j
        int k  = i % DIM;
        int gcol = (lc >> 4) * UNITS + ubase + (lc & 15);
        Wlds[i] = f2bf(W[k * GDIM + gcol]);
    }
    __syncthreads();

    // per-lane cell state: 8 (batch,unit) cells matching C-matrix layout
    float creg[8];
#pragma unroll
    for (int r = 0; r < 8; ++r) creg[r] = 0.0f;

    const int batch0 = 16 * w;                   // wave's batch tile

    for (int t = 0; t < TLEN; ++t) {
        const unsigned short* hprev = hbuf + (t & 1) * BATCH * UNITS;
        unsigned short*       hnext = hbuf + ((t + 1) & 1) * BATCH * UNITS;

        const float* xzt = xz + (size_t)t * BATCH * GDIM;   // contiguous slice

        // ---- init accumulators with xz slice (z = xz_t + h @ W) ----
        v8f acc[4];
#pragma unroll
        for (int g = 0; g < 4; ++g) {
            int cg = g * UNITS + ubase + nlan;
#pragma unroll
            for (int r = 0; r < 8; ++r) {
                int batch = batch0 + r + 8 * half;
                acc[g][r] = xzt[(size_t)batch * GDIM + cg];
            }
        }
        // prefetch next step's contiguous xz slice (global_prefetch_b8)
        if (t + 1 < TLEN) {
            const float* xzn = xz + (size_t)(t + 1) * BATCH * GDIM;
            __builtin_prefetch(&xzn[(size_t)(batch0 + nlan) * GDIM + ubase],
                               0, 0);
        }

        // ---- z += h_prev @ W(cols)  : 4 tiles, K = 256 ----
#pragma unroll
        for (int ks = 0; ks < 8; ++ks) {
            FragCast fa;
            const unsigned short* ar = hprev + (batch0 + nlan) * UNITS;
#pragma unroll
            for (int j = 0; j < 8; ++j) {
                int kk = 32 * ks + kmap(j, half);
                fa.u[j] = *reinterpret_cast<const unsigned int*>(ar + kk);
            }
#pragma unroll
            for (int g = 0; g < 4; ++g) {
                FragCast fb;
                const unsigned short* bcol = Wlds + (g * 16 + nlan) * DIM;
#pragma unroll
                for (int j = 0; j < 8; ++j) {
                    int kk = 32 * ks + kmap(j, half);
                    fb.u[j] = *reinterpret_cast<const unsigned int*>(bcol + kk);
                }
                acc[g] = __builtin_amdgcn_wmma_f32_16x16x32_bf16(
                    false, fa.b, false, fb.b, (short)0, acc[g], false, false);
            }
        }

        // ---- gates (entirely in-register, per (batch,unit) cell) ----
#pragma unroll
        for (int r = 0; r < 8; ++r) {
            float ig = fsigmoid(acc[0][r]);
            float fg = fsigmoid(acc[1][r]);
            float gg = ftanh(acc[2][r]);
            float og = fsigmoid(acc[3][r]);
            float c  = fg * creg[r] + ig * gg;
            creg[r]  = c;
            float h  = og * ftanh(c);

            int batch = batch0 + r + 8 * half;
            int unit  = ubase + nlan;
            hnext[batch * UNITS + unit] = f2bf(h);
            if (t >= TLEN - NOUT) {
                out[((size_t)batch * NOUT + (t - (TLEN - NOUT))) * UNITS + unit] = h;
            }
        }

        // ---- device-wide barrier (release/acquire) ----
        __threadfence();
        __syncthreads();
        if (tid == 0) {
            atomicAdd(cnt, 1u);
            unsigned int target = (unsigned int)NBLK_R * (unsigned int)(t + 1);
            while (*reinterpret_cast<volatile unsigned int*>(cnt) < target) {
                __builtin_amdgcn_s_sleep(1);
            }
        }
        __syncthreads();
        __threadfence();
    }
}

// ---------------------------------------------------------------------------

extern "C" void kernel_launch(void* const* d_in, const int* in_sizes, int n_in,
                              void* d_out, int out_size, void* d_ws, size_t ws_size,
                              hipStream_t stream) {
    (void)in_sizes; (void)n_in; (void)out_size; (void)ws_size;

    const float* x    = (const float*)d_in[0];   // [64,1024,256]
    const float* U    = (const float*)d_in[1];   // [256,1024]
    const float* W    = (const float*)d_in[2];   // [256,1024]
    const float* bias = (const float*)d_in[3];   // [1024]
    float* out = (float*)d_out;                  // [64,32,256]

    const size_t M = (size_t)BATCH * TLEN;

    // workspace layout
    char* base = (char*)d_ws;
    float*          xz   = (float*)base;                          // 256 MB
    size_t off = M * GDIM * sizeof(float);
    unsigned short* xb   = (unsigned short*)(base + off);         // 32 MB
    off += M * DIM * sizeof(unsigned short);
    unsigned short* ubT  = (unsigned short*)(base + off);         // 512 KB
    off += (size_t)GDIM * DIM * sizeof(unsigned short);
    unsigned short* hbuf = (unsigned short*)(base + off);         // 64 KB
    off += 2 * (size_t)BATCH * UNITS * sizeof(unsigned short);
    off = (off + 127) & ~(size_t)127;
    unsigned int*   cnt  = (unsigned int*)(base + off);

    int nx = (int)(M * DIM);
    conv_x_kernel<<<(nx + 255) / 256, 256, 0, stream>>>(x, xb, nx);
    conv_U_kernel<<<(GDIM * DIM + 255) / 256, 256, 0, stream>>>(U, ubT);
    init_state_kernel<<<32, 256, 0, stream>>>(hbuf, cnt);

    // xz = x @ U + b : (M/64) * (1024/128) blocks
    input_proj_kernel<<<(int)(M / 64) * (GDIM / 128), 256, 0, stream>>>(
        xb, ubT, bias, xz);

    recurrence_kernel<<<NBLK_R, 128, 0, stream>>>(xz, W, hbuf, cnt, out);
}
